// gtnet_Signal_66786741452952
// MI455X (gfx1250) — matrix operand order repaired and checked by reading, not compiled, not run on hardware
//
#include <hip/hip_runtime.h>

#define Nn 512
#define Bb 8
#define Tt 32
#define Cc 32
#define Dd 40
#define Ss 64
#define Ee 128
#define Oo 12
#define NSLAB (Bb*Tt)          // 256
#define SLABSZ (Cc*Nn)         // 16384 elements per (b,t) slab, layout [slab][c][n]

typedef __attribute__((ext_vector_type(16))) __bf16 v16bf;
typedef __attribute__((ext_vector_type(8)))  float  v8f;

// ---------- helpers ----------
__device__ __forceinline__ unsigned short f2bf(float x) {
  union { float f; unsigned u; } c; c.f = x;
  unsigned r = c.u + 0x7fffu + ((c.u >> 16) & 1u);   // RNE
  return (unsigned short)(r >> 16);
}
__device__ __forceinline__ float sigmoidf_(float x) { return 1.0f / (1.0f + expf(-x)); }

union BF16x16 { uint4 u[2]; v16bf v; };

// A fragment (16x32 bf16), rows from LDS tile sA[32][512]
__device__ __forceinline__ v16bf load_a_frag(const unsigned short* sA, int row, int k0, int khalf) {
  BF16x16 t;
  t.u[0] = *(const uint4*)(sA + row * Nn + k0 + 16 * khalf);       // K = k0+16h .. +7
  t.u[1] = *(const uint4*)(sA + row * Nn + k0 + 16 * khalf + 8);   // K = k0+16h+8 .. +7
  return t.v;
}
// B fragment (32x16 bf16) from column-major matrix (row-major [n][k])
__device__ __forceinline__ v16bf load_b_frag(const unsigned short* p) {
  BF16x16 t;
  t.u[0] = ((const uint4*)p)[0];
  t.u[1] = ((const uint4*)p)[1];
  return t.v;
}

// ---------- 1. node embeddings ----------
__global__ void k_node_embed(const float* __restrict__ emb1, const float* __restrict__ emb2,
                             const float* __restrict__ l1w, const float* __restrict__ l1b,
                             const float* __restrict__ l2w, const float* __restrict__ l2b,
                             const int* __restrict__ idx,
                             float* __restrict__ nv1, float* __restrict__ nv2) {
  int i = blockIdx.x * blockDim.x + threadIdx.x;
  if (i >= 2 * Nn * Dd) return;
  int which = i / (Nn * Dd);
  int r = i % (Nn * Dd);
  int n = r / Dd, d = r % Dd;
  const float* emb = which ? emb2 : emb1;
  const float* W   = which ? l2w  : l1w;
  const float* bb  = which ? l2b  : l1b;
  int src = idx[n];
  float acc = bb[d];
  for (int k = 0; k < Dd; ++k) acc += emb[src * Dd + k] * W[d * Dd + k];
  float v = tanhf(3.0f * acc);
  (which ? nv2 : nv1)[n * Dd + d] = v;
}

// ---------- 2. adjacency: tanh -> row softmax -> mask; write bf16 adj and adj^T ----------
__global__ void k_adj(const float* __restrict__ nv1, const float* __restrict__ nv2,
                      const float* __restrict__ maskA,
                      unsigned short* __restrict__ adj_bf,   // [v][w] row-major
                      unsigned short* __restrict__ adjT_bf)  // [w][v] row-major
{
  int n = blockIdx.x, tid = threadIdx.x;
  __shared__ float s1[Dd];
  __shared__ float row[Nn];
  __shared__ float red[256];
  if (tid < Dd) s1[tid] = nv1[n * Dd + tid];
  __syncthreads();
  for (int w = tid; w < Nn; w += 256) {
    float acc = 0.f;
    for (int k = 0; k < Dd; ++k) acc += s1[k] * nv2[w * Dd + k];
    row[w] = tanhf(3.0f * acc);
  }
  __syncthreads();
  float m = -1e30f;
  for (int w = tid; w < Nn; w += 256) m = fmaxf(m, row[w]);
  red[tid] = m; __syncthreads();
  for (int s = 128; s > 0; s >>= 1) { if (tid < s) red[tid] = fmaxf(red[tid], red[tid + s]); __syncthreads(); }
  m = red[0]; __syncthreads();
  float ssum = 0.f;
  for (int w = tid; w < Nn; w += 256) { float e = expf(row[w] - m); row[w] = e; ssum += e; }
  red[tid] = ssum; __syncthreads();
  for (int s = 128; s > 0; s >>= 1) { if (tid < s) red[tid] += red[tid + s]; __syncthreads(); }
  float inv = 1.0f / red[0];
  for (int w = tid; w < Nn; w += 256) {
    float a = row[w] * inv * maskA[n * Nn + w];
    unsigned short h = f2bf(a);
    adj_bf[n * Nn + w]  = h;
    adjT_bf[w * Nn + n] = h;
  }
}

// ---------- 3. tiny vectors ua=Wa@u, ub=Wb@u, wv=W@v for both mixprops ----------
__global__ void k_prep(const float* Wa1, const float* u1, const float* Wb1,
                       const float* W1,  const float* v1,
                       const float* Wa2, const float* u2, const float* Wb2,
                       const float* W2,  const float* v2, float* __restrict__ vecs) {
  int t = threadIdx.x; if (t >= 192) return;
  int grp = t >> 5, c = t & 31;
  const float* M; const float* vv;
  switch (grp) {
    case 0:  M = Wa1; vv = u1; break;
    case 1:  M = Wb1; vv = u1; break;
    case 2:  M = W1;  vv = v1; break;
    case 3:  M = Wa2; vv = u2; break;
    case 4:  M = Wb2; vv = u2; break;
    default: M = W2;  vv = v2; break;
  }
  float acc = 0.f;
  for (int d = 0; d < Cc; ++d) acc += M[c * Cc + d] * vv[d];
  vecs[grp * Cc + c] = acc;
}

// ---------- 4. start 1x1 conv -> h0 [slab][c][n] ----------
__global__ void k_start(const float* __restrict__ x, const float* __restrict__ sw,
                        const float* __restrict__ sb, float* __restrict__ h0) {
  int i = blockIdx.x * 256 + threadIdx.x;
  if (i >= NSLAB * SLABSZ) return;
  int n = i & (Nn - 1);
  int c = (i >> 9) & 31;
  int slab = i >> 14;
  int t = slab & 31, b = slab >> 5;
  float xv = x[(b * Nn + n) * Tt + t];
  h0[i] = xv * sw[c] + sb[c];
}

// ---------- 5. dilated inception TCN + tanh*sigmoid gate ----------
__global__ void k_tcn(const float* __restrict__ h0,
                      const float* fw3, const float* fw5, const float* fw7, const float* fw9,
                      const float* gw3, const float* gw5, const float* gw7, const float* gw9,
                      const float* __restrict__ f_b, const float* __restrict__ g_b,
                      float* __restrict__ xg, unsigned short* __restrict__ xg_bf) {
  __shared__ float sW[9 * Cc * Cc];   // 36 KB: combined tap table, zero-padded per branch
  int tid = threadIdx.x;
  int gidx = blockIdx.x * 256 + tid;          // (slab, n)
  int n = gidx & (Nn - 1), slab = gidx >> 9;
  int t = slab & 31, b = slab >> 5;
  const float* fws[4] = { fw3, fw5, fw7, fw9 };
  const float* gws[4] = { gw3, gw5, gw7, gw9 };
  float accF[Cc], accG[Cc];
  for (int pass = 0; pass < 2; ++pass) {
    __syncthreads();
    for (int e = tid; e < 9 * Cc * Cc; e += 256) {
      int co = e & 31, ci = (e >> 5) & 31, p = e >> 10;
      int br = co >> 3, coin = co & 7;
      int k = 3 + 2 * br, off = 3 - br;       // off = (8 - pad_k)/2, pad_k = k-1 (DIL=2)
      int j = p - off;
      float w = 0.f;
      if (j >= 0 && j < k) {
        const float* srcw = pass ? gws[br] : fws[br];
        w = srcw[(coin * Cc + ci) * k + j];
      }
      sW[e] = w;
    }
    __syncthreads();
    if (pass) { for (int c = 0; c < Cc; ++c) accG[c] = g_b[c]; }
    else      { for (int c = 0; c < Cc; ++c) accF[c] = f_b[c]; }
    for (int p = 0; p < 9; ++p) {
      int tt = t - 8 + 2 * p;
      if (tt < 0 || tt >= Tt) continue;
      const float* hb = h0 + (size_t)((b * Tt + tt) * Cc) * Nn + n;
      for (int ci = 0; ci < Cc; ++ci) {
        float val = hb[ci * Nn];
        const float* wr = &sW[(p * Cc + ci) * Cc];
#pragma unroll
        for (int co = 0; co < Cc; ++co) {
          float w = wr[co] * val;
          if (pass) accG[co] += w; else accF[co] += w;
        }
      }
    }
  }
  for (int c = 0; c < Cc; ++c) {
    float v = tanhf(accF[c]) * sigmoidf_(accG[c]);
    size_t o = (size_t)slab * SLABSZ + c * Nn + n;
    xg[o] = v;
    xg_bf[o] = f2bf(v);
  }
}

// ---------- 6. WMMA diffusion GEMM: Out[slab] = A[slab] (32x512 bf16) @ B (512x512 bf16) ----------
// Bcm is B stored column-major (row-major [n][k]); grid = (4 n-quarters, 256 slabs), 256 thr.
__global__ void k_gemm(const unsigned short* __restrict__ Abf,
                       const unsigned short* __restrict__ Bcm,
                       float* __restrict__ Out) {
  __shared__ __align__(16) unsigned short sA[Cc * Nn];   // 32 KB slab of A
  int slab = blockIdx.y, tid = threadIdx.x;
  { // cooperative stage: 2048 x uint4
    const uint4* src = (const uint4*)(Abf + (size_t)slab * SLABSZ);
    uint4* dst = (uint4*)sA;
    for (int i = tid; i < SLABSZ / 8; i += 256) dst[i] = src[i];
  }
  __syncthreads();
  int wave = tid >> 5, lane = tid & 31;
  int ntile = blockIdx.x * 8 + wave;
  int ncol  = ntile * 16 + (lane & 15);
  int khalf = lane >> 4;
  int row16 = lane & 15;
  v8f acc0 = {0.f,0.f,0.f,0.f,0.f,0.f,0.f,0.f};
  v8f acc1 = {0.f,0.f,0.f,0.f,0.f,0.f,0.f,0.f};
  const unsigned short* Bcol = Bcm + (size_t)ncol * Nn + 16 * khalf;
  for (int ks = 0; ks < 16; ++ks) {
    int k0 = ks * 32;
    v16bf a0 = load_a_frag(sA, row16,      k0, khalf);   // M rows 0..15
    v16bf a1 = load_a_frag(sA, 16 + row16, k0, khalf);   // M rows 16..31
    v16bf bb = load_b_frag(Bcol + k0);
    acc0 = __builtin_amdgcn_wmma_f32_16x16x32_bf16(false, a0, false, bb, (short)0, acc0, false, false);
    acc1 = __builtin_amdgcn_wmma_f32_16x16x32_bf16(false, a1, false, bb, (short)0, acc1, false, false);
  }
  float* out = Out + (size_t)slab * SLABSZ;
  int m0 = 8 * khalf;
#pragma unroll
  for (int r = 0; r < 8; ++r) {
    out[(m0 + r) * Nn + ncol]      = acc0[r];
    out[(16 + m0 + r) * Nn + ncol] = acc1[r];
  }
}

// ---------- 7. hop update: gate + mix-hop channel transform ----------
__global__ void k_hop(const float* __restrict__ hin, const float* __restrict__ hprev,
                      const float* hdiff, const float* __restrict__ vecs,
                      int off_a, int off_b,
                      const float* __restrict__ mw, const float* __restrict__ mb,
                      float* hnext, unsigned short* __restrict__ hnext_bf) {
  int slab = blockIdx.x, tid = threadIdx.x;
  __shared__ float smw[Cc * Cc];
  __shared__ float sua[Cc], sub[Cc], smb[Cc];
  for (int i = tid; i < Cc * Cc; i += 256) smw[i] = mw[i];
  if (tid < Cc) { sua[tid] = vecs[off_a + tid]; sub[tid] = vecs[off_b + tid]; smb[tid] = mb[tid]; }
  __syncthreads();
  const float* hinS = hin   + (size_t)slab * SLABSZ;
  const float* hpS  = hprev + (size_t)slab * SLABSZ;
  const float* hdS  = hdiff + (size_t)slab * SLABSZ;
  float* hoS        = hnext + (size_t)slab * SLABSZ;
  unsigned short* hbS = hnext_bf + (size_t)slab * SLABSZ;
  for (int n = tid; n < Nn; n += 256) {
    float hi[Cc], hm[Cc];
    float e_in = 0.f, e_l = 0.f;
    for (int c = 0; c < Cc; ++c) {
      hi[c] = hinS[c * Nn + n];
      float hp = hpS[c * Nn + n];
      e_in += hi[c] * sua[c];
      e_l  += hp    * sub[c];
    }
    float a = sigmoidf_(e_l - e_in);
    for (int c = 0; c < Cc; ++c) hm[c] = (1.f - a) * hi[c] + a * hdS[c * Nn + n];
    for (int c = 0; c < Cc; ++c) {
      float acc = smb[c];
      const float* wr = &smw[c * Cc];
      for (int d = 0; d < Cc; ++d) acc += wr[d] * hm[d];
      acc = fmaxf(acc, 0.f);
      hoS[c * Nn + n] = acc;
      hbS[c * Nn + n] = f2bf(acc);
    }
  }
}

// ---------- 8. hop combine with softmax-beta, plus residual/accumulate ----------
__global__ void k_combine(const float* __restrict__ hop0, const float* __restrict__ hop1,
                          const float* __restrict__ hop2, const float* __restrict__ vecs,
                          int off_wv, const float* addsrc, float* dst) {
  int slab = blockIdx.x, tid = threadIdx.x;
  __shared__ float swv[Cc];
  if (tid < Cc) swv[tid] = vecs[off_wv + tid];
  __syncthreads();
  const float* a0 = hop0 + (size_t)slab * SLABSZ;
  const float* a1 = hop1 + (size_t)slab * SLABSZ;
  const float* a2 = hop2 + (size_t)slab * SLABSZ;
  const float* ad = addsrc + (size_t)slab * SLABSZ;
  float* dd = dst + (size_t)slab * SLABSZ;
  for (int n = tid; n < Nn; n += 256) {
    float v0[Cc], v1[Cc], v2[Cc];
    float g0 = 0.f, g1 = 0.f, g2 = 0.f;
    for (int c = 0; c < Cc; ++c) {
      v0[c] = a0[c * Nn + n]; g0 += v0[c] * swv[c];
      v1[c] = a1[c * Nn + n]; g1 += v1[c] * swv[c];
      v2[c] = a2[c * Nn + n]; g2 += v2[c] * swv[c];
    }
    float m = fmaxf(g0, fmaxf(g1, g2));
    float e0 = expf(g0 - m), e1 = expf(g1 - m), e2 = expf(g2 - m);
    float inv = 1.0f / (e0 + e1 + e2);
    float b0 = e0 * inv, b1 = e1 * inv, b2 = e2 * inv;
    for (int c = 0; c < Cc; ++c) {
      float val = b0 * v0[c] + b1 * v1[c] + b2 * v2[c] + ad[c * Nn + n];
      dd[c * Nn + n] = val;
    }
  }
}

// ---------- 9. head: relu(skip_w@(xg+xr)+skip_b) -> relu(end1) -> end2 ----------
__global__ void k_head(const float* __restrict__ xg, const float* __restrict__ xr,
                       const float* skip_w, const float* skip_b,
                       const float* e1w, const float* e1b,
                       const float* e2w, const float* e2b,
                       float* __restrict__ out) {
  __shared__ float sskw[Ss * Cc];
  __shared__ float se1[Ee * Ss];
  __shared__ float se2[Oo * Ee];
  __shared__ float sskb[Ss], se1b[Ee], se2b[Oo];
  int tid = threadIdx.x;
  for (int i = tid; i < Ss * Cc; i += 256) sskw[i] = skip_w[i];
  for (int i = tid; i < Ee * Ss; i += 256) se1[i]  = e1w[i];
  for (int i = tid; i < Oo * Ee; i += 256) se2[i]  = e2w[i];
  if (tid < Ss) sskb[tid] = skip_b[tid];
  if (tid < Ee) se1b[tid] = e1b[tid];
  if (tid < Oo) se2b[tid] = e2b[tid];
  __syncthreads();
  int gidx = blockIdx.x * 256 + tid;          // (slab, n)
  int n = gidx & (Nn - 1), slab = gidx >> 9;
  int t = slab & 31, b = slab >> 5;
  float tcol[Cc];
  for (int c = 0; c < Cc; ++c) {
    size_t o = (size_t)slab * SLABSZ + c * Nn + n;
    tcol[c] = xg[o] + xr[o];
  }
  float sv[Ss];
  for (int s = 0; s < Ss; ++s) {
    float acc = sskb[s];
    const float* wr = &sskw[s * Cc];
    for (int c = 0; c < Cc; ++c) acc += wr[c] * tcol[c];
    sv[s] = fmaxf(acc, 0.f);
  }
  float ov[Oo];
#pragma unroll
  for (int o = 0; o < Oo; ++o) ov[o] = se2b[o];
  for (int e = 0; e < Ee; ++e) {
    float acc = se1b[e];
    const float* wr = &se1[e * Ss];
    for (int s = 0; s < Ss; ++s) acc += wr[s] * sv[s];
    acc = fmaxf(acc, 0.f);
#pragma unroll
    for (int o = 0; o < Oo; ++o) ov[o] += se2[o * Ee + e] * acc;
  }
  for (int o = 0; o < Oo; ++o)
    out[(((size_t)b * Oo + o) * Nn + n) * Tt + t] = ov[o];
}

// ---------------------------------------------------------------------------
extern "C" void kernel_launch(void* const* d_in, const int* in_sizes, int n_in,
                              void* d_out, int out_size, void* d_ws, size_t ws_size,
                              hipStream_t stream) {
  (void)in_sizes; (void)n_in; (void)out_size; (void)ws_size;
  const float* x      = (const float*)d_in[0];
  const int*   idx    = (const int*)  d_in[1];
  const float* maskA  = (const float*)d_in[2];
  const float* emb1   = (const float*)d_in[3];
  const float* emb2   = (const float*)d_in[4];
  const float* lin1_w = (const float*)d_in[5];
  const float* lin1_b = (const float*)d_in[6];
  const float* lin2_w = (const float*)d_in[7];
  const float* lin2_b = (const float*)d_in[8];
  const float* start_w= (const float*)d_in[9];
  const float* start_b= (const float*)d_in[10];
  const float* f_b    = (const float*)d_in[11];
  const float* g_b    = (const float*)d_in[12];
  const float* skip_w = (const float*)d_in[13];
  const float* skip_b = (const float*)d_in[14];
  const float* end1_w = (const float*)d_in[15];
  const float* end1_b = (const float*)d_in[16];
  const float* end2_w = (const float*)d_in[17];
  const float* end2_b = (const float*)d_in[18];
  const float* fw3 = (const float*)d_in[19];
  const float* gw3 = (const float*)d_in[20];
  const float* fw5 = (const float*)d_in[21];
  const float* gw5 = (const float*)d_in[22];
  const float* fw7 = (const float*)d_in[23];
  const float* gw7 = (const float*)d_in[24];
  const float* fw9 = (const float*)d_in[25];
  const float* gw9 = (const float*)d_in[26];
  const float* Wa1 = (const float*)d_in[27];
  const float* Wb1 = (const float*)d_in[28];
  const float* u1  = (const float*)d_in[29];
  const float* W1  = (const float*)d_in[30];
  const float* v1  = (const float*)d_in[31];
  const float* m1w = (const float*)d_in[32];
  const float* m1b = (const float*)d_in[33];
  const float* Wa2 = (const float*)d_in[34];
  const float* Wb2 = (const float*)d_in[35];
  const float* u2  = (const float*)d_in[36];
  const float* W2  = (const float*)d_in[37];
  const float* v2  = (const float*)d_in[38];
  const float* m2w = (const float*)d_in[39];
  const float* m2b = (const float*)d_in[40];

  // ---- workspace carve (256B aligned) ----
  size_t off = 0;
  char* base = (char*)d_ws;
  auto alloc = [&](size_t bytes) -> void* {
    void* p = base + off;
    off = (off + bytes + 255) & ~(size_t)255;
    return p;
  };
  float* nv1  = (float*)alloc((size_t)Nn * Dd * 4);
  float* nv2  = (float*)alloc((size_t)Nn * Dd * 4);
  float* vecs = (float*)alloc(192 * 4);
  unsigned short* adj_bf  = (unsigned short*)alloc((size_t)Nn * Nn * 2);  // adj row-major  (B-cm for mixprop2)
  unsigned short* adjT_bf = (unsigned short*)alloc((size_t)Nn * Nn * 2);  // adj^T row-major (B-cm for mixprop1)
  float* h0 = (float*)alloc((size_t)NSLAB * SLABSZ * 4);   // start-conv out; later aliased as hg (xr)
  float* xg = (float*)alloc((size_t)NSLAB * SLABSZ * 4);
  unsigned short* xg_bf = (unsigned short*)alloc((size_t)NSLAB * SLABSZ * 2);
  float* h1 = (float*)alloc((size_t)NSLAB * SLABSZ * 4);
  unsigned short* h1_bf = (unsigned short*)alloc((size_t)NSLAB * SLABSZ * 2);
  float* hdiff = (float*)alloc((size_t)NSLAB * SLABSZ * 4); // also reused as h2

  // ---- graph construction ----
  k_node_embed<<<(2 * Nn * Dd + 255) / 256, 256, 0, stream>>>(emb1, emb2, lin1_w, lin1_b,
                                                              lin2_w, lin2_b, idx, nv1, nv2);
  k_adj<<<Nn, 256, 0, stream>>>(nv1, nv2, maskA, adj_bf, adjT_bf);
  k_prep<<<1, 192, 0, stream>>>(Wa1, u1, Wb1, W1, v1, Wa2, u2, Wb2, W2, v2, vecs);

  // ---- start conv + TCN gate ----
  k_start<<<(NSLAB * SLABSZ) / 256, 256, 0, stream>>>(x, start_w, start_b, h0);
  k_tcn<<<(NSLAB * Nn) / 256, 256, 0, stream>>>(h0, fw3, fw5, fw7, fw9,
                                                gw3, gw5, gw7, gw9, f_b, g_b, xg, xg_bf);

  dim3 ggrid(4, NSLAB, 1);
  // ---- mixprop 1 (adj) ----
  k_gemm<<<ggrid, 256, 0, stream>>>(xg_bf, adjT_bf, hdiff);
  k_hop <<<NSLAB, 256, 0, stream>>>(xg, xg, hdiff, vecs, 0, 32, m1w, m1b, h1, h1_bf);
  k_gemm<<<ggrid, 256, 0, stream>>>(h1_bf, adjT_bf, hdiff);
  k_hop <<<NSLAB, 256, 0, stream>>>(xg, h1, hdiff, vecs, 0, 32, m1w, m1b, hdiff, h1_bf); // h2 in-place
  k_combine<<<NSLAB, 256, 0, stream>>>(xg, h1, hdiff, vecs, 64, h0, h0);  // hg = mix1 + h0

  // ---- mixprop 2 (adj^T) ----
  k_gemm<<<ggrid, 256, 0, stream>>>(xg_bf, adj_bf, hdiff);
  k_hop <<<NSLAB, 256, 0, stream>>>(xg, xg, hdiff, vecs, 96, 128, m2w, m2b, h1, h1_bf);
  k_gemm<<<ggrid, 256, 0, stream>>>(h1_bf, adj_bf, hdiff);
  k_hop <<<NSLAB, 256, 0, stream>>>(xg, h1, hdiff, vecs, 96, 128, m2w, m2b, hdiff, h1_bf);
  k_combine<<<NSLAB, 256, 0, stream>>>(xg, h1, hdiff, vecs, 160, h0, h0); // hg += mix2

  // ---- head ----
  k_head<<<(NSLAB * Nn) / 256, 256, 0, stream>>>(xg, h0, skip_w, skip_b,
                                                 end1_w, end1_b, end2_w, end2_b,
                                                 (float*)d_out);
}
